// SyncMixtralSparseMoeBlock_77884936945957
// MI455X (gfx1250) — matrix operand order repaired and checked by reading, not compiled, not run on hardware
//
#include <hip/hip_runtime.h>
#include <stdint.h>

#define T_TOK 8192
#define HID   2048
#define FFN_D 4096
#define NEXP  8

#define BM    128
#define BK    32
#define LDS_P 40   /* uint16 elements per LDS row: 32 + 8 pad (16B) */

#if defined(__has_builtin)
#  if __has_builtin(__builtin_amdgcn_tensor_load_to_lds)
#    define HAVE_TDM 1
#  endif
#endif
#ifndef HAVE_TDM
#  define HAVE_TDM 0
#endif

typedef __attribute__((ext_vector_type(16))) __bf16 v16bf;
typedef __attribute__((ext_vector_type(8)))  float  v8f;
typedef __attribute__((ext_vector_type(4)))  unsigned int v4u;
typedef __attribute__((ext_vector_type(8)))  int v8i_t;
typedef __attribute__((ext_vector_type(4)))  int v4i_t;

union ABFrag { v16bf v; uint4 q[2]; };

__device__ __forceinline__ uint32_t f2bf1(float f) {
  union { float f; uint32_t u; } cv; cv.f = f;
  uint32_t u = cv.u;
  u += 0x7FFFu + ((u >> 16) & 1u);   // round-to-nearest-even
  return u >> 16;
}
__device__ __forceinline__ uint32_t pack_bf2(float lo, float hi) {
  return f2bf1(lo) | (f2bf1(hi) << 16);
}
__device__ __forceinline__ uint2 pack_bf4(float4 f) {
  return make_uint2(pack_bf2(f.x, f.y), pack_bf2(f.z, f.w));
}

// Low 32 bits of a flat pointer into the LDS aperture == LDS byte offset.
__device__ __forceinline__ uint32_t lds_off(const void* p) {
  return (uint32_t)(uintptr_t)p;
}

#if HAVE_TDM
// TDM: load a 128(row) x 32(elem) bf16 tile (row stride FFN_D elems) into LDS
// with 16B padding after each 64B row (-> 80B LDS rows, matches LDS_P=40).
__device__ __forceinline__ void tdm_load_tile_128x32(uint32_t lds_byte,
                                                     const uint16_t* gtile) {
  const uint64_t ga = (uint64_t)(uintptr_t)gtile;
  v4u g0;
  g0.x = 1u;                                   // count=1 (valid descriptor)
  g0.y = lds_byte;                             // lds_addr
  g0.z = (uint32_t)ga;                         // global_addr[31:0]
  g0.w = (uint32_t)(ga >> 32) | (2u << 30);    // global_addr[56:32] | type=2
  v8i_t g1;
  g1[0] = (int)((1u << 16)      // data_size = 2 bytes
              | (1u << 20)      // pad_enable
              | (3u << 22)      // pad_interval: 16 DWORDs (one 64B row)
              | (3u << 25));    // pad_amount : 4 DWORDs (16B)
  g1[1] = (int)((FFN_D & 0xFFFFu) << 16);                  // tensor_dim0 lo16
  g1[2] = (int)((FFN_D >> 16) |
                (((2u * T_TOK) & 0xFFFFu) << 16));         // dim0 hi | dim1 lo
  g1[3] = (int)(((2u * T_TOK) >> 16) | (32u << 16));       // dim1 hi | tile_dim0=32
  g1[4] = 128;                                             // tile_dim1=128
  g1[5] = FFN_D;                                           // tensor_dim0_stride
  g1[6] = 0;
  g1[7] = 0;
  v4i_t z4 = {0, 0, 0, 0};
  v8i_t z8 = {0, 0, 0, 0, 0, 0, 0, 0};
  // 6-arg form (this toolchain): (g0, g1, g2, g3, extra, cpol)
  __builtin_amdgcn_tensor_load_to_lds(g0, g1, z4, z4, z8, 0);
}
#endif

// ---------------------------------------------------------------------------
// 1) Router: wave-per-token. logits = x @ gate_w.T ; top-2 + renormalize.
// ---------------------------------------------------------------------------
__global__ __launch_bounds__(256) void moe_router(
    const float* __restrict__ x, const float* __restrict__ gw,
    float* __restrict__ logits_out, int* __restrict__ topIdx,
    float* __restrict__ topW)
{
  const int lane = threadIdx.x & 31;
  const int wid  = threadIdx.x >> 5;
  const int t = blockIdx.x * 8 + wid;
  const float* xr = x + (size_t)t * HID;

  float acc[NEXP];
  #pragma unroll
  for (int e = 0; e < NEXP; ++e) acc[e] = 0.f;
  for (int h = lane; h < HID; h += 32) {
    float xv = xr[h];
    #pragma unroll
    for (int e = 0; e < NEXP; ++e) acc[e] += xv * gw[e * HID + h];
  }
  #pragma unroll
  for (int e = 0; e < NEXP; ++e) {
    #pragma unroll
    for (int m = 16; m >= 1; m >>= 1) acc[e] += __shfl_xor(acc[e], m, 32);
  }
  if (lane == 0) {
    float mx = acc[0];
    #pragma unroll
    for (int e = 1; e < NEXP; ++e) mx = fmaxf(mx, acc[e]);
    float p[NEXP];
    #pragma unroll
    for (int e = 0; e < NEXP; ++e) {
      logits_out[t * NEXP + e] = acc[e];
      p[e] = __expf(acc[e] - mx);   // softmax denom cancels in top-2 renorm
    }
    int i0 = 0; float v0 = p[0];
    #pragma unroll
    for (int e = 1; e < NEXP; ++e) if (p[e] > v0) { v0 = p[e]; i0 = e; }
    int i1 = (i0 == 0) ? 1 : 0; float v1 = p[i1];
    #pragma unroll
    for (int e = 0; e < NEXP; ++e) if (e != i0 && p[e] > v1) { v1 = p[e]; i1 = e; }
    float inv = __builtin_amdgcn_rcpf(v0 + v1);
    topIdx[2 * t]     = i0;
    topIdx[2 * t + 1] = i1;
    topW[2 * t]       = v0 * inv;
    topW[2 * t + 1]   = v1 * inv;
  }
}

// ---------------------------------------------------------------------------
// 2) Deterministic per-expert stream compaction (block-per-expert, LDS scan)
// ---------------------------------------------------------------------------
__global__ __launch_bounds__(256) void moe_gather(
    const int* __restrict__ topIdx, int* __restrict__ lists,
    int* __restrict__ counts)
{
  const int e = blockIdx.x;
  const int tid = threadIdx.x;
  __shared__ int s[256];
  int base = 0;
  for (int c0 = 0; c0 < T_TOK; c0 += 256) {
    const int t = c0 + tid;
    const int pred = (topIdx[2 * t] == e || topIdx[2 * t + 1] == e) ? 1 : 0;
    s[tid] = pred;
    __syncthreads();
    #pragma unroll
    for (int off = 1; off < 256; off <<= 1) {
      int add = (tid >= off) ? s[tid - off] : 0;
      __syncthreads();
      s[tid] += add;
      __syncthreads();
    }
    if (pred) lists[e * T_TOK + base + s[tid] - 1] = t;
    int tot = s[255];
    __syncthreads();
    base += tot;
  }
  if (tid == 0) counts[e] = base;
}

__global__ void moe_offsets(const int* __restrict__ counts, int* __restrict__ offs) {
  if (threadIdx.x == 0 && blockIdx.x == 0) {
    int a = 0;
    for (int e = 0; e < NEXP; ++e) { offs[e] = a; a += counts[e]; }
    offs[NEXP] = a;   // == 2*T_TOK
  }
}

__global__ __launch_bounds__(256) void moe_pairslot(
    const int* __restrict__ lists, const int* __restrict__ counts,
    const int* __restrict__ offs, const int* __restrict__ topIdx,
    int* __restrict__ pairSlot)
{
  const int idx = blockIdx.x * 256 + threadIdx.x;   // over NEXP*T_TOK
  const int e = idx / T_TOK, pos = idx % T_TOK;
  if (pos >= counts[e]) return;
  const int t = lists[idx];
  const int kk = (topIdx[2 * t] == e) ? 0 : 1;
  pairSlot[2 * t + kk] = offs[e] + pos;
}

// ---------------------------------------------------------------------------
// 5) FFN phase 1: gathered  h = silu(X@W1^T) * (X@W3^T)  -> bf16 hbuf
//    128x64x32 tile, double-buffered LDS, register-staged global loads.
// ---------------------------------------------------------------------------
__global__ __launch_bounds__(256) void moe_ffn1(
    const float* __restrict__ x, const float* __restrict__ w1,
    const float* __restrict__ w3, const int* __restrict__ lists,
    const int* __restrict__ counts, const int* __restrict__ offs,
    uint16_t* __restrict__ hbuf)
{
  const int e = blockIdx.z, mt = blockIdx.y, nt = blockIdx.x;
  const int Ne = counts[e];
  if (mt * BM >= Ne) return;
  const int slotbase = offs[e] + mt * BM;

  __shared__ int      sTok[BM];
  __shared__ uint16_t sA [2][BM * LDS_P];
  __shared__ uint16_t sB1[2][64 * LDS_P];
  __shared__ uint16_t sB3[2][64 * LDS_P];

  const int tid  = threadIdx.x;
  const int lane = tid & 31, wid = tid >> 5;
  const int wm = wid >> 1, wn = wid & 1;

  if (tid < BM) {
    int pos = mt * BM + tid;
    sTok[tid] = lists[e * T_TOK + (pos < Ne ? pos : Ne - 1)];
  }
  __syncthreads();

  const float* w1g = w1 + ((size_t)e * FFN_D + (size_t)nt * 64) * HID;
  const float* w3g = w3 + ((size_t)e * FFN_D + (size_t)nt * 64) * HID;

  // Per-thread source pointers / LDS offsets (k-invariant)
  const float* asrc[4]; int aoff[4];
  #pragma unroll
  for (int i = 0; i < 4; ++i) {
    int id = tid + i * 256;
    int row = id >> 3, c4 = (id & 7) << 2;
    asrc[i] = x + (size_t)sTok[row] * HID + c4;
    aoff[i] = row * LDS_P + c4;
  }
  const float* b1src[2]; const float* b3src[2]; int boff[2];
  #pragma unroll
  for (int i = 0; i < 2; ++i) {
    int id = tid + i * 256;
    int row = id >> 3, c4 = (id & 7) << 2;
    b1src[i] = w1g + (size_t)row * HID + c4;
    b3src[i] = w3g + (size_t)row * HID + c4;
    boff[i] = row * LDS_P + c4;
  }

  v8f c1[2][2] = {}, c3[2][2] = {};
  float4 ra[4], rb1[2], rb3[2];

  // Prologue: tile k=0
  #pragma unroll
  for (int i = 0; i < 4; ++i) ra[i]  = *(const float4*)(asrc[i]);
  #pragma unroll
  for (int i = 0; i < 2; ++i) { rb1[i] = *(const float4*)(b1src[i]);
                                rb3[i] = *(const float4*)(b3src[i]); }
  #pragma unroll
  for (int i = 0; i < 4; ++i) *(uint2*)&sA[0][aoff[i]]  = pack_bf4(ra[i]);
  #pragma unroll
  for (int i = 0; i < 2; ++i) { *(uint2*)&sB1[0][boff[i]] = pack_bf4(rb1[i]);
                                *(uint2*)&sB3[0][boff[i]] = pack_bf4(rb3[i]); }
  __syncthreads();

  const int NK = HID / BK;   // 64
  for (int kt = 0; kt < NK; ++kt) {
    const int cur = kt & 1, nxt = cur ^ 1;
    const int k1 = (kt + 1) * BK;
    const bool more = (kt + 1 < NK);
    if (more) {   // stage next tile into registers while WMMAs run
      #pragma unroll
      for (int i = 0; i < 4; ++i) ra[i]  = *(const float4*)(asrc[i] + k1);
      #pragma unroll
      for (int i = 0; i < 2; ++i) { rb1[i] = *(const float4*)(b1src[i] + k1);
                                    rb3[i] = *(const float4*)(b3src[i] + k1); }
    }

    ABFrag a[2], b1f[2], b3f[2];
    const int akoff = (lane >> 4) << 4;   // K chunk base: 0 / 16
    const int bkoff = (lane >> 4) << 4;   // K chunk base: 0 / 16
    const uint16_t* sAc  = sA[cur];
    const uint16_t* sB1c = sB1[cur];
    const uint16_t* sB3c = sB3[cur];
    #pragma unroll
    for (int mi = 0; mi < 2; ++mi) {
      const uint16_t* p = &sAc[(wm * 32 + mi * 16 + (lane & 15)) * LDS_P + akoff];
      a[mi].q[0] = *(const uint4*)p;
      a[mi].q[1] = *(const uint4*)(p + 8);
    }
    #pragma unroll
    for (int ni = 0; ni < 2; ++ni) {
      const uint16_t* p1 = &sB1c[(wn * 32 + ni * 16 + (lane & 15)) * LDS_P + bkoff];
      b1f[ni].q[0] = *(const uint4*)p1;
      b1f[ni].q[1] = *(const uint4*)(p1 + 8);
      const uint16_t* p3 = &sB3c[(wn * 32 + ni * 16 + (lane & 15)) * LDS_P + bkoff];
      b3f[ni].q[0] = *(const uint4*)p3;
      b3f[ni].q[1] = *(const uint4*)(p3 + 8);
    }
    #pragma unroll
    for (int mi = 0; mi < 2; ++mi)
      #pragma unroll
      for (int ni = 0; ni < 2; ++ni) {
        c1[mi][ni] = __builtin_amdgcn_wmma_f32_16x16x32_bf16(
            false, a[mi].v, false, b1f[ni].v, (short)0, c1[mi][ni], false, false);
        c3[mi][ni] = __builtin_amdgcn_wmma_f32_16x16x32_bf16(
            false, a[mi].v, false, b3f[ni].v, (short)0, c3[mi][ni], false, false);
      }

    if (more) {   // store staged tile into the other buffer
      #pragma unroll
      for (int i = 0; i < 4; ++i) *(uint2*)&sA[nxt][aoff[i]]  = pack_bf4(ra[i]);
      #pragma unroll
      for (int i = 0; i < 2; ++i) { *(uint2*)&sB1[nxt][boff[i]] = pack_bf4(rb1[i]);
                                    *(uint2*)&sB3[nxt][boff[i]] = pack_bf4(rb3[i]); }
    }
    __syncthreads();
  }

  // Fused SiLU * gate epilogue, store bf16
  #pragma unroll
  for (int mi = 0; mi < 2; ++mi)
    #pragma unroll
    for (int ni = 0; ni < 2; ++ni)
      #pragma unroll
      for (int r = 0; r < 8; ++r) {
        int row = wm * 32 + mi * 16 + ((lane >> 4) << 3) + r;
        if (mt * BM + row < Ne) {
          int col = nt * 64 + wn * 32 + ni * 16 + (lane & 15);
          float z1 = c1[mi][ni][r];
          float z3 = c3[mi][ni][r];
          float h = z1 * __builtin_amdgcn_rcpf(1.f + __expf(-z1)) * z3;
          hbuf[(size_t)(slotbase + row) * FFN_D + col] = (uint16_t)f2bf1(h);
        }
      }
}

// ---------------------------------------------------------------------------
// 6) FFN phase 2: pairOut = hbuf @ W2^T.  128x128x32 tile, double-buffered;
//    A tile (already bf16, contiguous rows) moved by the Tensor Data Mover.
// ---------------------------------------------------------------------------
__global__ __launch_bounds__(256) void moe_ffn2(
    const uint16_t* __restrict__ hbuf, const float* __restrict__ w2,
    const int* __restrict__ counts, const int* __restrict__ offs,
    float* __restrict__ pairOut)
{
  const int e = blockIdx.z, mt = blockIdx.y, nt = blockIdx.x;
  const int Ne = counts[e];
  if (mt * BM >= Ne) return;
  const int slot0 = offs[e];
  const int slotbase = slot0 + mt * BM;

  __shared__ uint16_t sA[2][BM * LDS_P];
  __shared__ uint16_t sB[2][BM * LDS_P];

  const int tid = threadIdx.x;
  const int lane = tid & 31, wid = tid >> 5;
  const int wm = wid >> 1, wn = wid & 1;

  v8f c[2][4] = {};

  const float* w2g = w2 + ((size_t)e * HID + (size_t)nt * 128) * FFN_D;
  const float* bsrc[4]; int boff[4];
  #pragma unroll
  for (int i = 0; i < 4; ++i) {
    int id = tid + i * 256;
    int row = id >> 3, c4 = (id & 7) << 2;
    bsrc[i] = w2g + (size_t)row * FFN_D + c4;
    boff[i] = row * LDS_P + c4;
  }
  float4 rb[4];

#if HAVE_TDM
  const uint16_t* agbase = hbuf + (size_t)slotbase * FFN_D;
#else
  const uint16_t* asrc[2]; int aoff[2];
  #pragma unroll
  for (int i = 0; i < 2; ++i) {
    int id = tid + i * 256;
    int row = id >> 2, c8 = (id & 3) << 3;
    int pos = mt * BM + row;
    int slot = slot0 + (pos < Ne ? pos : Ne - 1);
    asrc[i] = hbuf + (size_t)slot * FFN_D + c8;
    aoff[i] = row * LDS_P + c8;
  }
  uint4 raq[2];
#endif

  // Prologue: tile k=0
#if HAVE_TDM
  if (wid == 0) tdm_load_tile_128x32(lds_off(&sA[0][0]), agbase);
#else
  #pragma unroll
  for (int i = 0; i < 2; ++i) raq[i] = *(const uint4*)(asrc[i]);
  #pragma unroll
  for (int i = 0; i < 2; ++i) *(uint4*)&sA[0][aoff[i]] = raq[i];
#endif
  #pragma unroll
  for (int i = 0; i < 4; ++i) rb[i] = *(const float4*)(bsrc[i]);
  #pragma unroll
  for (int i = 0; i < 4; ++i) *(uint2*)&sB[0][boff[i]] = pack_bf4(rb[i]);
#if HAVE_TDM
  if (wid == 0) __builtin_amdgcn_s_wait_tensorcnt(0);
#endif
  __syncthreads();

  const int NK = FFN_D / BK;   // 128
  for (int kt = 0; kt < NK; ++kt) {
    const int cur = kt & 1, nxt = cur ^ 1;
    const int k1 = (kt + 1) * BK;
    const bool more = (kt + 1 < NK);
    if (more) {
#if HAVE_TDM
      if (wid == 0) tdm_load_tile_128x32(lds_off(&sA[nxt][0]), agbase + k1);
#else
      #pragma unroll
      for (int i = 0; i < 2; ++i) raq[i] = *(const uint4*)(asrc[i] + k1);
#endif
      #pragma unroll
      for (int i = 0; i < 4; ++i) rb[i] = *(const float4*)(bsrc[i] + k1);
    }

    ABFrag a[2], b[4];
    const int akoff = (lane >> 4) << 4;
    const int bkoff = (lane >> 4) << 4;
    const uint16_t* sAc = sA[cur];
    const uint16_t* sBc = sB[cur];
    #pragma unroll
    for (int mi = 0; mi < 2; ++mi) {
      const uint16_t* p = &sAc[(wm * 32 + mi * 16 + (lane & 15)) * LDS_P + akoff];
      a[mi].q[0] = *(const uint4*)p;
      a[mi].q[1] = *(const uint4*)(p + 8);
    }
    #pragma unroll
    for (int ni = 0; ni < 4; ++ni) {
      const uint16_t* p = &sBc[(wn * 64 + ni * 16 + (lane & 15)) * LDS_P + bkoff];
      b[ni].q[0] = *(const uint4*)p;
      b[ni].q[1] = *(const uint4*)(p + 8);
    }
    #pragma unroll
    for (int mi = 0; mi < 2; ++mi)
      #pragma unroll
      for (int ni = 0; ni < 4; ++ni)
        c[mi][ni] = __builtin_amdgcn_wmma_f32_16x16x32_bf16(
            false, a[mi].v, false, b[ni].v, (short)0, c[mi][ni], false, false);

    if (more) {
#if !HAVE_TDM
      #pragma unroll
      for (int i = 0; i < 2; ++i) *(uint4*)&sA[nxt][aoff[i]] = raq[i];
#endif
      #pragma unroll
      for (int i = 0; i < 4; ++i) *(uint2*)&sB[nxt][boff[i]] = pack_bf4(rb[i]);
    }
#if HAVE_TDM
    if (more && wid == 0) __builtin_amdgcn_s_wait_tensorcnt(0);
#endif
    __syncthreads();
  }

  #pragma unroll
  for (int mi = 0; mi < 2; ++mi)
    #pragma unroll
    for (int ni = 0; ni < 4; ++ni)
      #pragma unroll
      for (int r = 0; r < 8; ++r) {
        int row = wm * 32 + mi * 16 + ((lane >> 4) << 3) + r;
        if (mt * BM + row < Ne) {
          int col = nt * 128 + wn * 64 + ni * 16 + (lane & 15);
          pairOut[(size_t)(slotbase + row) * HID + col] = c[mi][ni][r];
        }
      }
}

// ---------------------------------------------------------------------------
// 7) Deterministic combine: out[t] = w0*pairOut[s0] + w1*pairOut[s1]
// ---------------------------------------------------------------------------
__global__ __launch_bounds__(256) void moe_combine(
    const float* __restrict__ pairOut, const int* __restrict__ pairSlot,
    const float* __restrict__ topW, float* __restrict__ outp)
{
  const int idx = blockIdx.x * 256 + threadIdx.x;   // over T*(H/4)
  const int t  = idx >> 9;                          // H/4 == 512
  const int c4 = (idx & 511) << 2;
  const int s0 = pairSlot[2 * t], s1 = pairSlot[2 * t + 1];
  const float w0 = topW[2 * t], w1v = topW[2 * t + 1];
  float4 a = *(const float4*)(pairOut + (size_t)s0 * HID + c4);
  float4 b = *(const float4*)(pairOut + (size_t)s1 * HID + c4);
  float4 o;
  o.x = w0 * a.x + w1v * b.x;
  o.y = w0 * a.y + w1v * b.y;
  o.z = w0 * a.z + w1v * b.z;
  o.w = w0 * a.w + w1v * b.w;
  *(float4*)(outp + (size_t)t * HID + c4) = o;
}

// ---------------------------------------------------------------------------
extern "C" void kernel_launch(void* const* d_in, const int* in_sizes, int n_in,
                              void* d_out, int out_size, void* d_ws, size_t ws_size,
                              hipStream_t stream)
{
  (void)in_sizes; (void)n_in; (void)out_size; (void)ws_size;
  const float* x  = (const float*)d_in[0];
  const float* gw = (const float*)d_in[1];
  const float* w1 = (const float*)d_in[2];
  const float* w3 = (const float*)d_in[3];
  const float* w2 = (const float*)d_in[4];

  float* outp = (float*)d_out;
  float* logits_out = outp + (size_t)T_TOK * HID;

  char* ws = (char*)d_ws;
  int*      topIdx   = (int*)(ws + 0);                          // 64 KiB
  float*    topW     = (float*)(ws + (64 << 10));               // 64 KiB
  int*      counts   = (int*)(ws + (128 << 10));
  int*      offs     = (int*)(ws + (132 << 10));
  int*      lists    = (int*)(ws + (136 << 10));                // 256 KiB
  int*      pairSlot = (int*)(ws + (512 << 10));                // 64 KiB
  uint16_t* hbuf     = (uint16_t*)(ws + (1 << 20));             // 128 MiB
  float*    pairOut  = (float*)(ws + (1 << 20) +
                                (size_t)2 * T_TOK * FFN_D * sizeof(uint16_t)); // 128 MiB

  moe_router  <<<T_TOK / 8, 256, 0, stream>>>(x, gw, logits_out, topIdx, topW);
  moe_gather  <<<NEXP, 256, 0, stream>>>(topIdx, lists, counts);
  moe_offsets <<<1, 32, 0, stream>>>(counts, offs);
  moe_pairslot<<<(NEXP * T_TOK) / 256, 256, 0, stream>>>(lists, counts, offs, topIdx, pairSlot);
  moe_ffn1    <<<dim3(FFN_D / 64, T_TOK / BM, NEXP), 256, 0, stream>>>(
                  x, w1, w3, lists, counts, offs, hbuf);
  moe_ffn2    <<<dim3(HID / 128, T_TOK / BM, NEXP), 256, 0, stream>>>(
                  hbuf, w2, counts, offs, pairOut);
  moe_combine <<<(T_TOK * (HID / 4)) / 256, 256, 0, stream>>>(
                  pairOut, pairSlot, topW, outp);
}
